// MultiHeadAttention_14757507629390
// MI455X (gfx1250) — matrix operand order repaired and checked by reading, not compiled, not run on hardware
//
#include <hip/hip_runtime.h>
#include <hip/hip_bf16.h>

// ---------------------------------------------------------------------------
// MHA forward for MI455X (gfx1250): f16 WMMA (f32 accumulate) everywhere,
// flash-attention streaming (no T x T materialization), L2-resident operands
// (192MB L2 >> ~50MB working set => no LDS staging of K/V; LDS used only for
// the C-layout -> A-layout exchange of the softmax P tile).
// Round 2: register blocking. Flash waves own 32 query rows (K/V fragments
// shared by two row tiles, 2x arithmetic intensity); projection waves own
// 16x64 output (A fragment shared by 4 B fragments); branch-free stores.
// ---------------------------------------------------------------------------

typedef _Float16 half_t;
typedef __attribute__((ext_vector_type(16))) _Float16 v16h;
typedef __attribute__((ext_vector_type(8)))  _Float16 v8h;
typedef __attribute__((ext_vector_type(8)))  float    v8f;

#define WMMA_F16(a, b, c) \
  __builtin_amdgcn_wmma_f32_16x16x32_f16(false, (a), false, (b), (short)0, (c), false, false)

// ---- fragment loaders (ISA 7.12.2 layouts, wave32) -------------------------
// A (16x32 f16, MxK): lane = 16h+m holds row m; halves 0..7 -> K=8h..8h+7,
// halves 8..15 -> K=16+8h..16+8h+7.
static __device__ __forceinline__ v16h load_a16(const half_t* __restrict__ base,
                                                int lda, int lane) {
  const int h = (lane >> 4) & 1;
  const int m = lane & 15;
  const half_t* p = base + (size_t)m * lda;
  const v8h lo = *(const v8h*)(p + 8 * h);
  const v8h hi = *(const v8h*)(p + 16 + 8 * h);
  v16h a;
#pragma unroll
  for (int i = 0; i < 8; ++i) { a[i] = lo[i]; a[i + 8] = hi[i]; }
  return a;
}

// B (32x16 f16, KxN): lane = 16h+n holds column n; halves 0..7 -> K=8h..8h+7,
// halves 8..15 -> K=16+8h..16+8h+7.  "base" points at (k=0, col n0); column n's K-run is
// contiguous in memory with stride ldb between columns.
static __device__ __forceinline__ v16h load_b16(const half_t* __restrict__ base,
                                                int ldb, int lane) {
  const int h = (lane >> 4) & 1;
  const int n = lane & 15;
  const half_t* p = base + (size_t)n * ldb;
  const v8h lo = *(const v8h*)(p + 8 * h);
  const v8h hi = *(const v8h*)(p + 16 + 8 * h);
  v16h b;
#pragma unroll
  for (int i = 0; i < 8; ++i) { b[i] = lo[i]; b[i + 8] = hi[i]; }
  return b;
}

// ---- kernel 0: f32 -> f16 conversion ---------------------------------------
__global__ void cvt_f32_to_f16(const float* __restrict__ src,
                               half_t* __restrict__ dst, int n) {
  int i = blockIdx.x * blockDim.x + threadIdx.x;
  const int stride = gridDim.x * blockDim.x;
  for (; i < n; i += stride) dst[i] = (half_t)src[i];
}

// ---- kernel 1: fused QKV projection ----------------------------------------
// out[m,n] = sum_k x[m,k] * W[n,k] + bias[n];  M=8192, N=512, K=512.
// gridDim = (64, 3): 128 rows/block (8 waves x 16), y selects Q/K/V.
// Wave computes 16x64 per N-step: one A fragment feeds 4 WMMAs.
// Q,K stored [b,h,t,d] f16; V stored transposed [b,h,d,t] f16.
__global__ __launch_bounds__(256) void qkv_proj(
    const half_t* __restrict__ xh,
    const half_t* __restrict__ Wqh, const half_t* __restrict__ Wkh,
    const half_t* __restrict__ Wvh,
    const float* __restrict__ bq, const float* __restrict__ bk,
    const float* __restrict__ bv,
    half_t* __restrict__ Qb, half_t* __restrict__ Kb, half_t* __restrict__ Vtb) {
  const int lane = threadIdx.x & 31;
  const int wave = threadIdx.x >> 5;
  const int m0   = blockIdx.x * 128 + wave * 16;
  const int sel  = blockIdx.y;
  const half_t* W    = (sel == 0) ? Wqh : (sel == 1) ? Wkh : Wvh;
  const float*  bias = (sel == 0) ? bq  : (sel == 1) ? bk  : bv;
  half_t* dstQK = (sel == 0) ? Qb : Kb;
  const int hp = (lane >> 4) & 1;
  const int nn = lane & 15;
  const half_t* arow = xh + (size_t)m0 * 512;

  for (int n0 = 0; n0 < 512; n0 += 64) {
    v8f acc[4];
    const v8f vz = {};
#pragma unroll
    for (int dt = 0; dt < 4; ++dt) acc[dt] = vz;
#pragma unroll 2
    for (int k0 = 0; k0 < 512; k0 += 32) {
      const v16h a = load_a16(arow + k0, 512, lane);
#pragma unroll
      for (int dt = 0; dt < 4; ++dt) {
        const v16h b = load_b16(W + (size_t)(n0 + dt * 16) * 512 + k0, 512, lane);
        acc[dt] = WMMA_F16(a, b, acc[dt]);
      }
    }
    if (sel != 2) {                     // Q or K: [b,h,t,d]
#pragma unroll
      for (int dt = 0; dt < 4; ++dt) {
        const int   ng = n0 + dt * 16 + nn;
        const float bs = bias[ng];
        const int   hh = ng >> 6, d = ng & 63;
#pragma unroll
        for (int r = 0; r < 8; ++r) {
          const int mg = m0 + r + 8 * hp;   // global row = b*T + t
          const int bb = mg >> 12, t = mg & 4095;
          dstQK[(((size_t)(bb * 8 + hh) << 12) + t) * 64 + d] =
              (half_t)(acc[dt][r] + bs);
        }
      }
    } else {                            // V: transposed [b,h,d,t]
#pragma unroll
      for (int dt = 0; dt < 4; ++dt) {
        const int   ng = n0 + dt * 16 + nn;
        const float bs = bias[ng];
        const int   hh = ng >> 6, d = ng & 63;
#pragma unroll
        for (int r = 0; r < 8; ++r) {
          const int mg = m0 + r + 8 * hp;
          const int bb = mg >> 12, t = mg & 4095;
          Vtb[(((size_t)(bb * 8 + hh) * 64 + d) << 12) + t] =
              (half_t)(acc[dt][r] + bs);
        }
      }
    }
  }
}

// ---- kernel 2: causal flash attention --------------------------------------
// grid = (T/256, B*H), block = 256 (8 waves).  Each wave: 32 query rows (two
// 16-row tiles A/B sharing all K/V fragments), streams 32-wide key tiles with
// online softmax; P goes through a per-wave LDS region to convert C-layout ->
// A-layout (same-wave DS is in-order, no barrier needed).
__global__ __launch_bounds__(256) void flash_attn(
    const half_t* __restrict__ Qb, const half_t* __restrict__ Kb,
    const half_t* __restrict__ Vtb, half_t* __restrict__ attn) {
  __shared__ __align__(16) half_t pshare[8][32 * 32];   // 16 KB
  const int lane = threadIdx.x & 31;
  const int wave = threadIdx.x >> 5;
  const int bh   = blockIdx.y;
  const int qt0  = blockIdx.x * 256 + wave * 32;
  const half_t* Qp = Qb + (size_t)bh * 4096 * 64;
  const half_t* Kp = Kb + (size_t)bh * 4096 * 64;
  const half_t* Vp = Vtb + (size_t)bh * 64 * 4096;
  const int hp = (lane >> 4) & 1;
  const int nn = lane & 15;

  // Q fragments for both 16-row tiles (Dh = 64 -> two K-steps each)
  const v16h aqA0 = load_a16(Qp + (size_t)qt0 * 64, 64, lane);
  const v16h aqA1 = load_a16(Qp + (size_t)qt0 * 64 + 32, 64, lane);
  const v16h aqB0 = load_a16(Qp + (size_t)(qt0 + 16) * 64, 64, lane);
  const v16h aqB1 = load_a16(Qp + (size_t)(qt0 + 16) * 64 + 32, 64, lane);

  float mA[8], lA[8], mB[8], lB[8];
  v8f   oA[4], oB[4];
  const v8f vz = {};
#pragma unroll
  for (int r = 0; r < 8; ++r) {
    mA[r] = -3.0e38f; lA[r] = 0.0f;
    mB[r] = -3.0e38f; lB[r] = 0.0f;
  }
#pragma unroll
  for (int dt = 0; dt < 4; ++dt) { oA[dt] = vz; oB[dt] = vz; }

  half_t* sp = &pshare[wave][0];

  for (int j0 = 0; j0 <= qt0; j0 += 32) {  // covers keys j <= qt0+31 = t_max
    if (j0 + 32 <= qt0) {                  // warm caches for next key tile
      __builtin_prefetch(Kp + (size_t)(j0 + 32) * 64, 0, 1);
      __builtin_prefetch(Vp + (size_t)(j0 + 32), 0, 1);
    }
    const bool diag = (j0 == qt0);         // only tile intersecting diagonal

    // shared K fragments for this key tile
    const v16h bk0  = load_b16(Kp + (size_t)j0 * 64, 64, lane);
    const v16h bk0h = load_b16(Kp + (size_t)j0 * 64 + 32, 64, lane);
    const v16h bk1  = load_b16(Kp + (size_t)(j0 + 16) * 64, 64, lane);
    const v16h bk1h = load_b16(Kp + (size_t)(j0 + 16) * 64 + 32, 64, lane);

    // ---- row-tile A: t in [qt0, qt0+16) ----
    {
      v8f s0 = {}, s1 = {};
      s0 = WMMA_F16(aqA0, bk0, s0);  s0 = WMMA_F16(aqA1, bk0h, s0);
      s1 = WMMA_F16(aqA0, bk1, s1);  s1 = WMMA_F16(aqA1, bk1h, s1);
      float p0[8], p1[8];
#pragma unroll
      for (int r = 0; r < 8; ++r) { p0[r] = s0[r] * 0.125f; p1[r] = s1[r] * 0.125f; }
      if (diag) {
#pragma unroll
        for (int r = 0; r < 8; ++r) {
          const int t = qt0 + r + 8 * hp;
          if (j0 + nn > t)      p0[r] = -3.0e38f;
          if (j0 + 16 + nn > t) p1[r] = -3.0e38f;
        }
      }
#pragma unroll
      for (int r = 0; r < 8; ++r) {
        float v = fmaxf(p0[r], p1[r]);
        v = fmaxf(v, __shfl_xor(v, 1, 32));
        v = fmaxf(v, __shfl_xor(v, 2, 32));
        v = fmaxf(v, __shfl_xor(v, 4, 32));
        v = fmaxf(v, __shfl_xor(v, 8, 32));
        const float nm   = fmaxf(mA[r], v);
        const float corr = __expf(mA[r] - nm);
        p0[r] = __expf(p0[r] - nm);
        p1[r] = __expf(p1[r] - nm);
        float sum = p0[r] + p1[r];
        sum += __shfl_xor(sum, 1, 32);
        sum += __shfl_xor(sum, 2, 32);
        sum += __shfl_xor(sum, 4, 32);
        sum += __shfl_xor(sum, 8, 32);
        lA[r] = lA[r] * corr + sum;
        mA[r] = nm;
#pragma unroll
        for (int dt = 0; dt < 4; ++dt) oA[dt][r] *= corr;
      }
#pragma unroll
      for (int r = 0; r < 8; ++r) {
        const int row = r + 8 * hp;
        sp[row * 32 + nn]      = (half_t)p0[r];
        sp[row * 32 + 16 + nn] = (half_t)p1[r];
      }
    }
    // ---- row-tile B: t in [qt0+16, qt0+32) ----
    {
      v8f s0 = {}, s1 = {};
      s0 = WMMA_F16(aqB0, bk0, s0);  s0 = WMMA_F16(aqB1, bk0h, s0);
      s1 = WMMA_F16(aqB0, bk1, s1);  s1 = WMMA_F16(aqB1, bk1h, s1);
      float p0[8], p1[8];
#pragma unroll
      for (int r = 0; r < 8; ++r) { p0[r] = s0[r] * 0.125f; p1[r] = s1[r] * 0.125f; }
      if (diag) {
#pragma unroll
        for (int r = 0; r < 8; ++r) {
          const int t = qt0 + 16 + r + 8 * hp;
          if (j0 + nn > t)      p0[r] = -3.0e38f;
          if (j0 + 16 + nn > t) p1[r] = -3.0e38f;
        }
      }
#pragma unroll
      for (int r = 0; r < 8; ++r) {
        float v = fmaxf(p0[r], p1[r]);
        v = fmaxf(v, __shfl_xor(v, 1, 32));
        v = fmaxf(v, __shfl_xor(v, 2, 32));
        v = fmaxf(v, __shfl_xor(v, 4, 32));
        v = fmaxf(v, __shfl_xor(v, 8, 32));
        const float nm   = fmaxf(mB[r], v);
        const float corr = __expf(mB[r] - nm);
        p0[r] = __expf(p0[r] - nm);
        p1[r] = __expf(p1[r] - nm);
        float sum = p0[r] + p1[r];
        sum += __shfl_xor(sum, 1, 32);
        sum += __shfl_xor(sum, 2, 32);
        sum += __shfl_xor(sum, 4, 32);
        sum += __shfl_xor(sum, 8, 32);
        lB[r] = lB[r] * corr + sum;
        mB[r] = nm;
#pragma unroll
        for (int dt = 0; dt < 4; ++dt) oB[dt][r] *= corr;
      }
#pragma unroll
      for (int r = 0; r < 8; ++r) {
        const int row = 16 + r + 8 * hp;
        sp[row * 32 + nn]      = (half_t)p0[r];
        sp[row * 32 + 16 + nn] = (half_t)p1[r];
      }
    }

    // read P back in A-layout (lda = 32) for both tiles
    v16h apA, apB;
    {
      const int m = lane & 15;
      const v8h loA = *(const v8h*)&pshare[wave][m * 32 + 8 * hp];
      const v8h hiA = *(const v8h*)&pshare[wave][m * 32 + 16 + 8 * hp];
      const v8h loB = *(const v8h*)&pshare[wave][(16 + m) * 32 + 8 * hp];
      const v8h hiB = *(const v8h*)&pshare[wave][(16 + m) * 32 + 16 + 8 * hp];
#pragma unroll
      for (int i = 0; i < 8; ++i) {
        apA[i] = loA[i]; apA[i + 8] = hiA[i];
        apB[i] = loB[i]; apB[i + 8] = hiB[i];
      }
    }

    // O += P * V : each V fragment feeds both row tiles
#pragma unroll
    for (int dt = 0; dt < 4; ++dt) {
      const v16h bv = load_b16(Vp + (size_t)(dt * 16) * 4096 + j0, 4096, lane);
      oA[dt] = WMMA_F16(apA, bv, oA[dt]);
      oB[dt] = WMMA_F16(apB, bv, oB[dt]);
    }
  }

  // finalize: O / l, store f16 as [b, t, h*64+d] (row-major 8192 x 512)
  const int b = bh >> 3, h = bh & 7;
  float iA[8], iB[8];
#pragma unroll
  for (int r = 0; r < 8; ++r) { iA[r] = 1.0f / lA[r]; iB[r] = 1.0f / lB[r]; }
#pragma unroll
  for (int dt = 0; dt < 4; ++dt) {
#pragma unroll
    for (int r = 0; r < 8; ++r) {
      const int tA = qt0 + r + 8 * hp;
      const int tB = qt0 + 16 + r + 8 * hp;
      const int dcol = h * 64 + dt * 16 + nn;
      attn[((size_t)(b * 4096 + tA)) * 512 + dcol] = (half_t)(oA[dt][r] * iA[r]);
      attn[((size_t)(b * 4096 + tB)) * 512 + dcol] = (half_t)(oB[dt][r] * iB[r]);
    }
  }
}

// ---- kernel 3: output projection (fp32 result) -----------------------------
__global__ __launch_bounds__(256) void out_proj(
    const half_t* __restrict__ ah, const half_t* __restrict__ Woh,
    const float* __restrict__ bo, float* __restrict__ out) {
  const int lane = threadIdx.x & 31;
  const int wave = threadIdx.x >> 5;
  const int m0   = blockIdx.x * 128 + wave * 16;
  const int hp = (lane >> 4) & 1;
  const int nn = lane & 15;
  const half_t* arow = ah + (size_t)m0 * 512;
  for (int n0 = 0; n0 < 512; n0 += 64) {
    v8f acc[4];
    const v8f vz = {};
#pragma unroll
    for (int dt = 0; dt < 4; ++dt) acc[dt] = vz;
#pragma unroll 2
    for (int k0 = 0; k0 < 512; k0 += 32) {
      const v16h a = load_a16(arow + k0, 512, lane);
#pragma unroll
      for (int dt = 0; dt < 4; ++dt) {
        const v16h b = load_b16(Woh + (size_t)(n0 + dt * 16) * 512 + k0, 512, lane);
        acc[dt] = WMMA_F16(a, b, acc[dt]);
      }
    }
#pragma unroll
    for (int dt = 0; dt < 4; ++dt) {
      const float bs = bo[n0 + dt * 16 + nn];
#pragma unroll
      for (int r = 0; r < 8; ++r)
        out[(size_t)(m0 + r + 8 * hp) * 512 + n0 + dt * 16 + nn] = acc[dt][r] + bs;
    }
  }
}

// ---------------------------------------------------------------------------
extern "C" void kernel_launch(void* const* d_in, const int* in_sizes, int n_in,
                              void* d_out, int out_size, void* d_ws, size_t ws_size,
                              hipStream_t stream) {
  const float* x  = (const float*)d_in[0];
  // d_in[1] = causal mask (int): unused, causality computed analytically
  const float* Wq = (const float*)d_in[2];
  const float* bq = (const float*)d_in[3];
  const float* Wk = (const float*)d_in[4];
  const float* bk = (const float*)d_in[5];
  const float* Wv = (const float*)d_in[6];
  const float* bv = (const float*)d_in[7];
  const float* Wo = (const float*)d_in[8];
  const float* bo = (const float*)d_in[9];
  float* out = (float*)d_out;

  const size_t NX = (size_t)8192 * 512;  // 4,194,304 elems
  const size_t NW = (size_t)512 * 512;   //   262,144 elems
  half_t* ws  = (half_t*)d_ws;           // total: 4*NX + 4*NW halves (~34 MB)
  half_t* xh    = ws;                    // x in f16; reused as attn buffer
  half_t* Wqh   = ws + NX;
  half_t* Wkh   = Wqh + NW;
  half_t* Wvh   = Wkh + NW;
  half_t* Woh   = Wvh + NW;
  half_t* Qb    = Woh + NW;              // [b,h,t,d]
  half_t* Kb    = Qb + NX;               // [b,h,t,d]
  half_t* Vtb   = Kb + NX;               // [b,h,d,t]
  half_t* attnb = xh;                    // alias: xh dead after qkv_proj

  cvt_f32_to_f16<<<2048, 256, 0, stream>>>(x,  xh,  (int)NX);
  cvt_f32_to_f16<<<256,  256, 0, stream>>>(Wq, Wqh, (int)NW);
  cvt_f32_to_f16<<<256,  256, 0, stream>>>(Wk, Wkh, (int)NW);
  cvt_f32_to_f16<<<256,  256, 0, stream>>>(Wv, Wvh, (int)NW);
  cvt_f32_to_f16<<<256,  256, 0, stream>>>(Wo, Woh, (int)NW);

  qkv_proj<<<dim3(64, 3), 256, 0, stream>>>(xh, Wqh, Wkh, Wvh, bq, bk, bv,
                                            Qb, Kb, Vtb);
  flash_attn<<<dim3(16, 16), 256, 0, stream>>>(Qb, Kb, Vtb, attnb);
  out_proj<<<dim3(64, 1), 256, 0, stream>>>(attnb, Woh, bo, out);
}
